// ResidualQuantizer_85023172591986
// MI455X (gfx1250) — matrix operand order repaired and checked by reading, not compiled, not run on hardware
//
#include <hip/hip_runtime.h>
#include <hip/hip_bf16.h>
#include <math.h>

// ---------------------------------------------------------------------------
// Residual VQ forward for MI455X (gfx1250).
// Distance GEMM: v_wmma_f32_16x16x32_bf16, codebook staged via
// global_load_async_to_lds_b128 (ASYNCcnt double buffering) into LDS.
// ---------------------------------------------------------------------------

typedef __bf16 bf16_t;
typedef __attribute__((ext_vector_type(16))) __bf16 v16bf;
typedef __attribute__((ext_vector_type(8)))  float  v8f;
typedef __attribute__((ext_vector_type(4)))  int    v4i_t;

#define B_  8
#define S_  2048
#define D_  256
#define L_  8
#define K_  2048
#define N_  (B_ * S_)          // 16384 rows
#define CSPLIT 4               // code-range splits per level (grid.y)
#define ROWS_PER_BLK 128       // 8 waves x 16 rows

#define CHUNK   32             // codes staged in LDS per step (2 tiles)
#define ROWB    512            // bytes per code row in global (256 bf16)
#define CROWB   528            // padded LDS row: 132 dwords -> conflict-free
#define NCHUNKS ((K_ / CSPLIT) / CHUNK)          // 16
#define CP_UNITS ((CHUNK * (ROWB / 16)) / 256)   // 16B units per thread = 4

union FragA { v16bf v; uint4 q[2]; };

// ---- async global->LDS path (guarded; falls back to sync copy) ------------
#if __has_builtin(__builtin_amdgcn_global_load_async_to_lds_b128)
#define CP_ASYNC(gp, lp)                                                     \
    __builtin_amdgcn_global_load_async_to_lds_b128((v4i_t*)(gp), (v4i_t*)(lp), 0, 0)
#if __has_builtin(__builtin_amdgcn_s_wait_asynccnt)
#define CP_WAIT(n) __builtin_amdgcn_s_wait_asynccnt(n)
#else
#define CP_WAIT(n) asm volatile("s_wait_asynccnt %0" ::"i"(n))
#endif
#else
#define CP_ASYNC(gp, lp) (*(uint4*)(lp) = *(const uint4*)(gp))
#define CP_WAIT(n)
#endif

// ---------------------------------------------------------------------------
// Convert codebooks f32 -> bf16 and compute |c|^2 per code. grid = L*K rows.
// ---------------------------------------------------------------------------
__global__ void rvq_prep_codebook(const float* __restrict__ cb,
                                  bf16_t* __restrict__ cb16,
                                  float* __restrict__ cnorm) {
    const int row = blockIdx.x;            // 0 .. L*K-1
    const int d   = threadIdx.x;           // 0 .. 255
    const size_t off = (size_t)row * D_ + d;
    const float v = cb[off];
    cb16[off] = (bf16_t)v;
    float s = v * v;
    #pragma unroll
    for (int m = 16; m >= 1; m >>= 1) s += __shfl_xor(s, m, 32);
    __shared__ float wsum[8];
    const int wave = threadIdx.x >> 5, lane = threadIdx.x & 31;
    if (lane == 0) wsum[wave] = s;
    __syncthreads();
    if (threadIdx.x == 0) {
        float t = 0.f;
        #pragma unroll
        for (int i = 0; i < 8; ++i) t += wsum[i];
        cnorm[row] = t;
    }
}

// ---------------------------------------------------------------------------
// residual_f32 = x ; residual_bf16 = bf16(x) ; qsum (d_out) = 0
// ---------------------------------------------------------------------------
__global__ void rvq_init_state(const float* __restrict__ x,
                               float* __restrict__ rf,
                               bf16_t* __restrict__ rb,
                               float* __restrict__ qout) {
    const size_t total = (size_t)N_ * D_;
    for (size_t i = (size_t)blockIdx.x * blockDim.x + threadIdx.x; i < total;
         i += (size_t)gridDim.x * blockDim.x) {
        const float v = x[i];
        rf[i] = v;
        rb[i] = (bf16_t)v;
        qout[i] = 0.f;
    }
}

__global__ void rvq_zero_misc(float* __restrict__ counts, float* __restrict__ loss) {
    const int i = blockIdx.x * blockDim.x + threadIdx.x;
    if (i < L_ * K_) counts[i] = 0.f;
    if (i < L_)      loss[i]   = 0.f;
}

// ---------------------------------------------------------------------------
// Fused distance GEMM + argmin for one level and one code split.
// grid = (N/128, CSPLIT), block = 256 (8 waves). Wave owns 16 rows.
// Codebook chunks staged in LDS (async, double buffered); two 16-code
// tiles run on independent accumulators to keep the WMMA pipe busy.
// ---------------------------------------------------------------------------
__global__ void __launch_bounds__(256)
rvq_dist_argmin(const bf16_t* __restrict__ rb,
                const bf16_t* __restrict__ cb16,  // level slice
                const float*  __restrict__ cnorm, // level slice
                float* __restrict__ pmin,
                int*   __restrict__ pidx) {
    __shared__ __align__(16) unsigned char smem[2 * CHUNK * CROWB];  // 33 KB

    const int wave = threadIdx.x >> 5;
    const int lane = threadIdx.x & 31;
    const int half = lane >> 4;            // 0: lanes 0-15, 1: lanes 16-31
    const int m    = lane & 15;
    const int rowBase   = blockIdx.x * ROWS_PER_BLK + wave * 16;
    const int splitBase = blockIdx.y * (K_ / CSPLIT);

    // Cache the wave's 16x256 residual panel as 8 WMMA A-fragments.
    FragA afrag[8];
    const bf16_t* rbase = rb + (size_t)(rowBase + m) * D_;
    #pragma unroll
    for (int ks = 0; ks < 8; ++ks) {
        const bf16_t* p = rbase + ks * 32 + half * 8;
        afrag[ks].q[0] = *(const uint4*)p;            // 8 bf16
        afrag[ks].q[1] = *(const uint4*)(p + 16);     // 8 bf16
    }

    // Stage chunk 0 (each thread copies CP_UNITS 16B units).
    {
        const int chunkBase = splitBase;
        #pragma unroll
        for (int u = 0; u < CP_UNITS; ++u) {
            const int unit = threadIdx.x + u * 256;   // 0..1023
            const int r    = unit >> 5;               // code row in chunk
            const int col  = unit & 31;               // 16B unit in row
            CP_ASYNC(cb16 + (size_t)(chunkBase + r) * D_ + col * 8,
                     smem + r * CROWB + col * 16);
        }
    }

    float minv[8];
    int   mini[8];
    #pragma unroll
    for (int v = 0; v < 8; ++v) { minv[v] = 3.4e38f; mini[v] = 0x7fffffff; }

    for (int c = 0; c < NCHUNKS; ++c) {
        // Prefetch next chunk into the other buffer, then retire current.
        if (c + 1 < NCHUNKS) {
            unsigned char* bufn = smem + ((c + 1) & 1) * (CHUNK * CROWB);
            const int chunkBase = splitBase + (c + 1) * CHUNK;
            #pragma unroll
            for (int u = 0; u < CP_UNITS; ++u) {
                const int unit = threadIdx.x + u * 256;
                const int r    = unit >> 5;
                const int col  = unit & 31;
                CP_ASYNC(cb16 + (size_t)(chunkBase + r) * D_ + col * 8,
                         bufn + r * CROWB + col * 16);
            }
            CP_WAIT(CP_UNITS);       // current chunk's copies are done
        } else {
            CP_WAIT(0);
        }
        __syncthreads();             // LDS chunk visible to all 8 waves

        const unsigned char* bufc = smem + (c & 1) * (CHUNK * CROWB);
        // B fragment rows: tile0 -> code m, tile1 -> code 16+m of the chunk.
        const unsigned char* brow0 = bufc + (size_t)m * CROWB + half * 16;
        const unsigned char* brow1 = brow0 + 16 * CROWB;

        v8f acc0 = {0.f, 0.f, 0.f, 0.f, 0.f, 0.f, 0.f, 0.f};
        v8f acc1 = {0.f, 0.f, 0.f, 0.f, 0.f, 0.f, 0.f, 0.f};
        #pragma unroll
        for (int ks = 0; ks < 8; ++ks) {
            FragA b0, b1;
            b0.q[0] = *(const uint4*)(brow0 + ks * 64);
            b0.q[1] = *(const uint4*)(brow0 + ks * 64 + 32);
            b1.q[0] = *(const uint4*)(brow1 + ks * 64);
            b1.q[1] = *(const uint4*)(brow1 + ks * 64 + 32);
            acc0 = __builtin_amdgcn_wmma_f32_16x16x32_bf16(
                false, afrag[ks].v, false, b0.v, (short)0, acc0, false, false);
            acc1 = __builtin_amdgcn_wmma_f32_16x16x32_bf16(
                false, afrag[ks].v, false, b1.v, (short)0, acc1, false, false);
        }

        const int  code0 = splitBase + c * CHUNK + m;   // N = lane&15
        const int  code1 = code0 + 16;
        const float cn0  = cnorm[code0];
        const float cn1  = cnorm[code1];
        #pragma unroll
        for (int v = 0; v < 8; ++v) {                   // M = v + 8*half
            const float d0 = cn0 - 2.0f * acc0[v];
            const float d1 = cn1 - 2.0f * acc1[v];
            if (d0 < minv[v]) { minv[v] = d0; mini[v] = code0; }
            if (d1 < minv[v]) { minv[v] = d1; mini[v] = code1; }
        }
        __syncthreads();             // all waves done reading this buffer
    }

    // Reduce min over the 16 lanes sharing each row M (butterfly per half).
    #pragma unroll
    for (int v = 0; v < 8; ++v) {
        #pragma unroll
        for (int mask = 1; mask <= 8; mask <<= 1) {
            const float ov = __shfl_xor(minv[v], mask, 32);
            const int   oi = __shfl_xor(mini[v], mask, 32);
            if (ov < minv[v] || (ov == minv[v] && oi < mini[v])) {
                minv[v] = ov; mini[v] = oi;
            }
        }
    }
    if (m == 0) {
        #pragma unroll
        for (int v = 0; v < 8; ++v) {
            const int row = rowBase + half * 8 + v;
            pmin[(size_t)blockIdx.y * N_ + row] = minv[v];
            pidx[(size_t)blockIdx.y * N_ + row] = mini[v];
        }
    }
}

// ---------------------------------------------------------------------------
// Combine split argmins, gather q (f32), update residual / qsum / loss /
// counts, emit index. grid = N rows, block = 256 = D.
// ---------------------------------------------------------------------------
__global__ void rvq_update_level(const float* __restrict__ cb_all, int level,
                                 float* __restrict__ rf, bf16_t* __restrict__ rb,
                                 float* __restrict__ qout,
                                 const float* __restrict__ pmin,
                                 const int*   __restrict__ pidx,
                                 float* __restrict__ counts,
                                 float* __restrict__ loss,
                                 float* __restrict__ idx_out) {
    const int row = blockIdx.x;
    float bv = pmin[row];
    int   bi = pidx[row];
    #pragma unroll
    for (int s = 1; s < CSPLIT; ++s) {
        const float v = pmin[(size_t)s * N_ + row];
        const int   i = pidx[(size_t)s * N_ + row];
        if (v < bv || (v == bv && i < bi)) { bv = v; bi = i; }
    }
    const int d = threadIdx.x;
    const size_t off = (size_t)row * D_ + d;
    const float q  = cb_all[((size_t)level * K_ + bi) * D_ + d];
    const float rn = rf[off] - q;
    rf[off]   = rn;
    rb[off]   = (bf16_t)rn;
    qout[off] += q;

    float s2 = rn * rn;
    #pragma unroll
    for (int mask = 16; mask >= 1; mask >>= 1) s2 += __shfl_xor(s2, mask, 32);
    __shared__ float wsum[8];
    const int wave = threadIdx.x >> 5, lane = threadIdx.x & 31;
    if (lane == 0) wsum[wave] = s2;
    __syncthreads();
    if (threadIdx.x == 0) {
        float t = 0.f;
        #pragma unroll
        for (int i = 0; i < 8; ++i) t += wsum[i];
        atomicAdd(&loss[level], t);
        atomicAdd(&counts[(size_t)level * K_ + bi], 1.0f);
        idx_out[(size_t)row * L_ + level] = (float)bi;
    }
}

// ---------------------------------------------------------------------------
// Perplexities + scaled commitment loss. Single block.
// ---------------------------------------------------------------------------
__global__ void rvq_finalize(const float* __restrict__ counts,
                             const float* __restrict__ loss,
                             float* __restrict__ out) {
    __shared__ float wsum[8];
    const float invn = 1.0f / (float)N_;
    const size_t baseStats = (size_t)N_ * D_ + (size_t)N_ * L_;
    for (int l = 0; l < L_; ++l) {
        float s = 0.f;
        for (int k = threadIdx.x; k < K_; k += 256) {
            const float p = counts[l * K_ + k] * invn;
            s += p * logf(p + 1e-10f);
        }
        #pragma unroll
        for (int mask = 16; mask >= 1; mask >>= 1) s += __shfl_xor(s, mask, 32);
        const int wave = threadIdx.x >> 5, lane = threadIdx.x & 31;
        if (lane == 0) wsum[wave] = s;
        __syncthreads();
        if (threadIdx.x == 0) {
            float t = 0.f;
            #pragma unroll
            for (int i = 0; i < 8; ++i) t += wsum[i];
            out[baseStats + 1 + l] = expf(-t);
        }
        __syncthreads();
    }
    if (threadIdx.x == 0) {
        float t = 0.f;
        #pragma unroll
        for (int l = 0; l < L_; ++l) t += loss[l];
        out[baseStats] = 0.25f * (t / (float)((size_t)N_ * D_));
    }
}

// ---------------------------------------------------------------------------
extern "C" void kernel_launch(void* const* d_in, const int* in_sizes, int n_in,
                              void* d_out, int out_size, void* d_ws, size_t ws_size,
                              hipStream_t stream) {
    const float* x  = (const float*)d_in[0];   // (B,S,D) f32
    const float* cb = (const float*)d_in[1];   // (L,K,D) f32
    float* out = (float*)d_out;

    // ---- workspace carve (256B aligned regions) ----
    char* ws = (char*)d_ws;
    float*  rf     = (float*) (ws + 0);              // N*D f32       (16 MB)
    bf16_t* rb     = (bf16_t*)(ws + 16777216);       // N*D bf16      ( 8 MB)
    bf16_t* cb16   = (bf16_t*)(ws + 25165824);       // L*K*D bf16    ( 8 MB)
    float*  cnorm  = (float*) (ws + 33554432);       // L*K f32       (64 KB)
    float*  counts = (float*) (ws + 33619968);       // L*K f32       (64 KB)
    float*  loss   = (float*) (ws + 33685504);       // L f32 (+pad)
    float*  pmin   = (float*) (ws + 33685760);       // CSPLIT*N f32  (256 KB)
    int*    pidx   = (int*)   (ws + 33947904);       // CSPLIT*N i32  (256 KB)

    float* qout    = out;                            // quantized region (qsum)
    float* idx_out = out + (size_t)N_ * D_;          // indices-as-float region

    rvq_prep_codebook<<<L_ * K_, 256, 0, stream>>>(cb, cb16, cnorm);
    rvq_init_state<<<4096, 256, 0, stream>>>(x, rf, rb, qout);
    rvq_zero_misc<<<(L_ * K_ + 255) / 256, 256, 0, stream>>>(counts, loss);

    for (int l = 0; l < L_; ++l) {
        dim3 grid(N_ / ROWS_PER_BLK, CSPLIT);
        rvq_dist_argmin<<<grid, 256, 0, stream>>>(
            rb, cb16 + (size_t)l * K_ * D_, cnorm + (size_t)l * K_, pmin, pidx);
        rvq_update_level<<<N_, 256, 0, stream>>>(
            cb, l, rf, rb, qout, pmin, pidx, counts, loss, idx_out);
    }

    rvq_finalize<<<1, 256, 0, stream>>>(counts, loss, out);
}
